// SelfAttention_63883343561108
// MI455X (gfx1250) — compile-verified, matrix operand order checked by reading, not run on hardware
//
#include <hip/hip_runtime.h>
#include <hip/hip_bf16.h>

// Problem constants (from reference)
#define Kdim  1024
#define Hn    16
#define HDim  64
#define Bn    2
#define Tn    2048
#define Mrows (Bn * Tn)   // 4096

typedef __bf16 bf16_t;
typedef __attribute__((ext_vector_type(16))) __bf16 bf16x16;
typedef __attribute__((ext_vector_type(8)))  __bf16 bf16x8;
typedef __attribute__((ext_vector_type(8)))  float  f32x8;

#define AS_GLOBAL __attribute__((address_space(1)))
#define AS_LDS    __attribute__((address_space(3)))

// Pointee type required by the async-LDS builtins (per hipcc diagnostic):
// pointer-to int __attribute__((vector_size(16))) in AS1 (src) / AS3 (dst).
typedef int v4i_vs __attribute__((vector_size(4 * sizeof(int))));

#if __has_builtin(__builtin_amdgcn_global_load_async_to_lds_b128) && \
    __has_builtin(__builtin_amdgcn_s_wait_asynccnt)
#define ASYNC_OK 1
#else
#define ASYNC_OK 0
#endif

// 16-byte global -> LDS copy. Async (ASYNCcnt-tracked) when the toolchain
// exposes the gfx1250 builtin, otherwise synchronous VGPR round-trip.
__device__ __forceinline__ void cp16_g2l(const bf16_t* g, bf16_t* l) {
#if ASYNC_OK
  __builtin_amdgcn_global_load_async_to_lds_b128(
      (AS_GLOBAL v4i_vs*)g, (AS_LDS v4i_vs*)l, 0, 0);
#else
  *(bf16x8*)l = *(const bf16x8*)g;
#endif
}

__device__ __forceinline__ void async_wait_all() {
#if ASYNC_OK
  __builtin_amdgcn_s_wait_asynccnt(0);
#endif
}

// D = A(16x32 bf16) * B(32x16 bf16) + C(16x16 f32), wave32 WMMA
__device__ __forceinline__ f32x8 wmma_bf16(bf16x16 a, bf16x16 b, f32x8 c) {
  return __builtin_amdgcn_wmma_f32_16x16x32_bf16(
      false, a, false, b, (short)0, c, false, false);
}

// Load one 16x32 bf16 A/B fragment (ISA 16-bit layout):
// lane l,hi=0: elems0-7 = K 0..7, elems8-15 = K 16..23
// lane l,hi=1: elems0-7 = K 8..15, elems8-15 = K 24..31
// Two contiguous 16B loads per lane (global_load_b128 or ds_load_b128).
__device__ __forceinline__ bf16x16 load_frag(const bf16_t* base, int stride,
                                             int row, int kbase, int hi) {
  const bf16_t* p = base + (size_t)row * stride + kbase + hi * 8;
  union { bf16x16 v; bf16x8 h[2]; } u;
  u.h[0] = *(const bf16x8*)(p);
  u.h[1] = *(const bf16x8*)(p + 16);
  return u.v;
}

__global__ void cvt_f32_bf16(const float* __restrict__ s, bf16_t* __restrict__ d, long n) {
  long i = (long)blockIdx.x * blockDim.x + threadIdx.x;
  if (i < n) d[i] = (bf16_t)s[i];
}

// ---------------------------------------------------------------------------
// Shared GEMM core: C(128x64) per 256-thread block (8 waves, 16x64 each),
// K stepped by 32 with double-buffered async LDS staging.
// LDS rows padded to 40 elems (80B) -> ds_load_b128 bank starts are exactly
// 4-spaced across 16 lanes: conflict-free.
// ---------------------------------------------------------------------------
__device__ __forceinline__ void stage_ab(const bf16_t* __restrict__ Ag,
                                         const bf16_t* __restrict__ Wg,
                                         bf16_t (&As)[128][40], bf16_t (&Bs)[64][40],
                                         int mblk, int n0, int kk, int tid) {
  // A tile 128x32 = 512 16B-chunks (2/thread), B tile 64x32 = 256 chunks (1/thread)
#pragma unroll
  for (int r = 0; r < 2; ++r) {
    int c = tid + r * 256;
    int row = c >> 2, cg = c & 3;
    cp16_g2l(Ag + (size_t)(mblk + row) * Kdim + kk + cg * 8, &As[row][cg * 8]);
  }
  {
    int row = tid >> 2, cg = tid & 3;
    cp16_g2l(Wg + (size_t)(n0 + row) * Kdim + kk + cg * 8, &Bs[row][cg * 8]);
  }
}

__device__ __forceinline__ void gemm_tile_core(const bf16_t* __restrict__ Ag,
                                               const bf16_t* __restrict__ Wg,
                                               bf16_t (&As)[2][128][40],
                                               bf16_t (&Bs)[2][64][40],
                                               int mblk, int n0, f32x8 (&acc)[4]) {
  const int tid  = threadIdx.x;
  const int wave = tid >> 5, lane = tid & 31;
  const int l = lane & 15, hi = lane >> 4;

  for (int nt = 0; nt < 4; ++nt)
    for (int i = 0; i < 8; ++i) acc[nt][i] = 0.f;

  stage_ab(Ag, Wg, As[0], Bs[0], mblk, n0, 0, tid);
  int cur = 0;
  for (int kk = 0; kk < Kdim; kk += 32) {
    async_wait_all();
    __syncthreads();
    if (kk + 32 < Kdim)
      stage_ab(Ag, Wg, As[cur ^ 1], Bs[cur ^ 1], mblk, n0, kk + 32, tid);

    bf16x16 a = load_frag(&As[cur][wave * 16][0], 40, l, 0, hi);
#pragma unroll
    for (int nt = 0; nt < 4; ++nt) {
      bf16x16 b = load_frag(&Bs[cur][nt * 16][0], 40, l, 0, hi);
      acc[nt] = wmma_bf16(a, b, acc[nt]);
    }
    cur ^= 1;
  }
}

// Fused Q/K/V projection: grid = (Kdim/64, Mrows/128, 3), block = 256.
// z=0 -> q (b,h,t,hd); z=1 -> k (b,h,t,hd); z=2 -> v transposed (b,h,hd,t).
__global__ __launch_bounds__(256) void gemm_qkv_kernel(
    const bf16_t* __restrict__ xb,
    const bf16_t* __restrict__ wqb, const bf16_t* __restrict__ wkb,
    const bf16_t* __restrict__ wvb,
    bf16_t* __restrict__ qb, bf16_t* __restrict__ kb, bf16_t* __restrict__ vtb) {
  __shared__ bf16_t As[2][128][40];
  __shared__ bf16_t Bs[2][64][40];
  const int wave = threadIdx.x >> 5;
  const int lane = threadIdx.x & 31;
  const int l = lane & 15, hi = lane >> 4;
  const int mblk = blockIdx.y * 128;
  const int n0   = blockIdx.x * 64;
  const int z    = blockIdx.z;
  const bf16_t* W = (z == 0) ? wqb : (z == 1) ? wkb : wvb;

  f32x8 acc[4];
  gemm_tile_core(xb, W, As, Bs, mblk, n0, acc);

  const int m0 = mblk + wave * 16;
  const int bi = m0 >> 11;
  const int t0 = m0 & (Tn - 1);
  if (z < 2) {
    bf16_t* dst = (z == 0) ? qb : kb;
#pragma unroll
    for (int nt = 0; nt < 4; ++nt) {
      int n = n0 + nt * 16 + l;
      int h = n >> 6, d = n & 63;
      bf16_t* p = dst + ((size_t)(bi * Hn + h) * Tn + t0 + 8 * hi) * HDim + d;
#pragma unroll
      for (int g = 0; g < 8; ++g) p[(size_t)g * HDim] = (bf16_t)acc[nt][g];
    }
  } else {
    // v^T: lane owns a column (fixed d) -> one packed 16B store per n-tile
#pragma unroll
    for (int nt = 0; nt < 4; ++nt) {
      int n = n0 + nt * 16 + l;
      int h = n >> 6, d = n & 63;
      bf16x8 pk;
#pragma unroll
      for (int g = 0; g < 8; ++g) pk[g] = (bf16_t)acc[nt][g];
      *(bf16x8*)(vtb + ((size_t)(bi * Hn + h) * HDim + d) * Tn + t0 + 8 * hi) = pk;
    }
  }
}

// ---------------------------------------------------------------------------
// Flash attention: grid = (Tn/64, Hn, Bn), block = 128 (4 waves, 16 q-rows
// each). K/V 32-token blocks are shared by all 4 waves -> async staged into
// double-buffered LDS. P tile relaid C->A via wave-private LDS.
// ---------------------------------------------------------------------------
__device__ __forceinline__ void stage_kv(const bf16_t* __restrict__ kptr,
                                         const bf16_t* __restrict__ vptr,
                                         bf16_t (&Ks)[32][72], bf16_t (&Vs)[64][40],
                                         int j0, int tid) {
  // K block 32x64 = 256 chunks (2/thread), V^T block 64x32 = 256 chunks (2/thread)
#pragma unroll
  for (int r = 0; r < 2; ++r) {
    int c = tid + r * 128;
    int row = c >> 3, cg = c & 7;
    cp16_g2l(kptr + (size_t)(j0 + row) * HDim + cg * 8, &Ks[row][cg * 8]);
  }
#pragma unroll
  for (int r = 0; r < 2; ++r) {
    int c = tid + r * 128;
    int row = c >> 2, cg = c & 3;
    cp16_g2l(vptr + (size_t)row * Tn + j0 + cg * 8, &Vs[row][cg * 8]);
  }
}

__global__ __launch_bounds__(128) void attn_kernel(
    const bf16_t* __restrict__ qg, const bf16_t* __restrict__ kg,
    const bf16_t* __restrict__ vtg, bf16_t* __restrict__ og) {
  __shared__ bf16_t Ks[2][32][72];
  __shared__ bf16_t Vs[2][64][40];
  __shared__ bf16_t plds[4][16][32];
  const int tid  = threadIdx.x;
  const int wave = tid >> 5;
  const int lane = tid & 31;
  const int l = lane & 15, hi = lane >> 4;
  const int h = blockIdx.y, bi = blockIdx.z;
  const int t0 = blockIdx.x * 64 + wave * 16;
  const float scale = 0.03125f;  // 1/sqrt(1024), full model dim per reference

  const bf16_t* qptr = qg  + (size_t)(bi * Hn + h) * Tn * HDim;
  const bf16_t* kptr = kg  + (size_t)(bi * Hn + h) * Tn * HDim;
  const bf16_t* vptr = vtg + (size_t)(bi * Hn + h) * HDim * Tn;

  bf16x16 qa0 = load_frag(qptr, HDim, t0 + l, 0, hi);
  bf16x16 qa1 = load_frag(qptr, HDim, t0 + l, 32, hi);

  f32x8 acc[4];
  for (int nt = 0; nt < 4; ++nt)
    for (int i = 0; i < 8; ++i) acc[nt][i] = 0.f;
  float mrow[8], lrow[8];
  for (int g = 0; g < 8; ++g) { mrow[g] = -1e30f; lrow[g] = 0.f; }

  stage_kv(kptr, vptr, Ks[0], Vs[0], 0, tid);
  int cur = 0;
  for (int j0 = 0; j0 < Tn; j0 += 32) {
    async_wait_all();
    __syncthreads();
    if (j0 + 32 < Tn)
      stage_kv(kptr, vptr, Ks[cur ^ 1], Vs[cur ^ 1], j0 + 32, tid);

    f32x8 s0, s1;
    for (int i = 0; i < 8; ++i) { s0[i] = 0.f; s1[i] = 0.f; }
    {
      bf16x16 k00 = load_frag(&Ks[cur][0][0], 72, l, 0, hi);
      bf16x16 k01 = load_frag(&Ks[cur][0][0], 72, l, 32, hi);
      s0 = wmma_bf16(qa0, k00, s0);
      s0 = wmma_bf16(qa1, k01, s0);
      bf16x16 k10 = load_frag(&Ks[cur][0][0], 72, 16 + l, 0, hi);
      bf16x16 k11 = load_frag(&Ks[cur][0][0], 72, 16 + l, 32, hi);
      s1 = wmma_bf16(qa0, k10, s1);
      s1 = wmma_bf16(qa1, k11, s1);
    }
    // Online softmax. C-layout: VGPR g = row (g+8*hi); shfl_xor masks 1..8
    // stay inside each 16-lane column group.
#pragma unroll
    for (int g = 0; g < 8; ++g) {
      float a = s0[g] * scale;
      float c = s1[g] * scale;
      float mx = fmaxf(a, c);
#pragma unroll
      for (int mk = 1; mk < 16; mk <<= 1) mx = fmaxf(mx, __shfl_xor(mx, mk, 32));
      float mnew  = fmaxf(mrow[g], mx);
      float alpha = __expf(mrow[g] - mnew);
      float p0 = __expf(a - mnew);
      float p1 = __expf(c - mnew);
      float rs = p0 + p1;
#pragma unroll
      for (int mk = 1; mk < 16; mk <<= 1) rs += __shfl_xor(rs, mk, 32);
      lrow[g] = lrow[g] * alpha + rs;
      mrow[g] = mnew;
      acc[0][g] *= alpha; acc[1][g] *= alpha;
      acc[2][g] *= alpha; acc[3][g] *= alpha;
      plds[wave][g + 8 * hi][l]      = (bf16_t)p0;
      plds[wave][g + 8 * hi][16 + l] = (bf16_t)p1;
    }
    // P back in A-layout (wave-private region, DScnt handled by compiler)
    bf16x16 pa = load_frag(&plds[wave][0][0], 32, l, 0, hi);
#pragma unroll
    for (int nt = 0; nt < 4; ++nt) {
      bf16x16 vb = load_frag(&Vs[cur][nt * 16][0], 40, l, 0, hi);
      acc[nt] = wmma_bf16(pa, vb, acc[nt]);
    }
    cur ^= 1;
  }

#pragma unroll
  for (int nt = 0; nt < 4; ++nt) {
    bf16_t* dst = og + ((size_t)bi * Tn + t0 + 8 * hi) * Kdim + h * HDim + nt * 16 + l;
#pragma unroll
    for (int g = 0; g < 8; ++g)
      dst[(size_t)g * Kdim] = (bf16_t)(acc[nt][g] / lrow[g]);
  }
}

// Output projection: out = ob @ Wu^T + bu (f32). grid = (16, 32), block = 256.
__global__ __launch_bounds__(256) void gemm_out_kernel(
    const bf16_t* __restrict__ ob, const bf16_t* __restrict__ wub,
    const float* __restrict__ bu, float* __restrict__ out) {
  __shared__ bf16_t As[2][128][40];
  __shared__ bf16_t Bs[2][64][40];
  const int wave = threadIdx.x >> 5;
  const int lane = threadIdx.x & 31;
  const int l = lane & 15, hi = lane >> 4;
  const int mblk = blockIdx.y * 128;
  const int n0   = blockIdx.x * 64;

  f32x8 acc[4];
  gemm_tile_core(ob, wub, As, Bs, mblk, n0, acc);

  const int m0 = mblk + wave * 16;
#pragma unroll
  for (int nt = 0; nt < 4; ++nt) {
    int n = n0 + nt * 16 + l;
    float bias = bu[n];
    float* dst = out + ((size_t)m0 + 8 * hi) * Kdim + n;
#pragma unroll
    for (int g = 0; g < 8; ++g) dst[(size_t)g * Kdim] = acc[nt][g] + bias;
  }
}

extern "C" void kernel_launch(void* const* d_in, const int* in_sizes, int n_in,
                              void* d_out, int out_size, void* d_ws, size_t ws_size,
                              hipStream_t stream) {
  (void)in_sizes; (void)n_in; (void)out_size; (void)ws_size;
  const float* x  = (const float*)d_in[0];
  const float* Wq = (const float*)d_in[1];
  const float* Wk = (const float*)d_in[2];
  const float* Wv = (const float*)d_in[3];
  const float* Wu = (const float*)d_in[4];
  const float* bu = (const float*)d_in[5];
  float* out = (float*)d_out;

  // Workspace layout (48 MB):
  // xb 8MB | wqb/wkb/wvb/wub 4x2MB | qb 8MB | kb 8MB | vtb 8MB | ob 8MB
  char* ws = (char*)d_ws;
  const size_t SZ_X = (size_t)Mrows * Kdim * sizeof(bf16_t);
  const size_t SZ_W = (size_t)Kdim * Kdim * sizeof(bf16_t);
  bf16_t* xb  = (bf16_t*)(ws);
  bf16_t* wqb = (bf16_t*)(ws + SZ_X);
  bf16_t* wkb = (bf16_t*)(ws + SZ_X + 1 * SZ_W);
  bf16_t* wvb = (bf16_t*)(ws + SZ_X + 2 * SZ_W);
  bf16_t* wub = (bf16_t*)(ws + SZ_X + 3 * SZ_W);
  bf16_t* qb  = (bf16_t*)(ws + 1 * SZ_X + 4 * SZ_W);
  bf16_t* kb  = (bf16_t*)(ws + 2 * SZ_X + 4 * SZ_W);
  bf16_t* vtb = (bf16_t*)(ws + 3 * SZ_X + 4 * SZ_W);
  bf16_t* ob  = (bf16_t*)(ws + 4 * SZ_X + 4 * SZ_W);

  const long nX = (long)Mrows * Kdim;
  const long nW = (long)Kdim * Kdim;
  cvt_f32_bf16<<<dim3((unsigned)((nX + 255) / 256)), dim3(256), 0, stream>>>(x,  xb,  nX);
  cvt_f32_bf16<<<dim3((unsigned)((nW + 255) / 256)), dim3(256), 0, stream>>>(Wq, wqb, nW);
  cvt_f32_bf16<<<dim3((unsigned)((nW + 255) / 256)), dim3(256), 0, stream>>>(Wk, wkb, nW);
  cvt_f32_bf16<<<dim3((unsigned)((nW + 255) / 256)), dim3(256), 0, stream>>>(Wv, wvb, nW);
  cvt_f32_bf16<<<dim3((unsigned)((nW + 255) / 256)), dim3(256), 0, stream>>>(Wu, wub, nW);

  gemm_qkv_kernel<<<dim3(Kdim / 64, Mrows / 128, 3), dim3(256), 0, stream>>>(
      xb, wqb, wkb, wvb, qb, kb, vtb);

  attn_kernel<<<dim3(Tn / 64, Hn, Bn), dim3(128), 0, stream>>>(qb, kb, vtb, ob);

  gemm_out_kernel<<<dim3(Kdim / 64, Mrows / 128, 1), dim3(256), 0, stream>>>(
      ob, wub, bu, out);
}
